// HAttention_27865747816877
// MI455X (gfx1250) — compile-verified, matrix operand order checked by reading, not run on hardware
//
#include <hip/hip_runtime.h>
#include <math.h>

// Problem constants (match reference)
#define Bn   16
#define Tn   2048
#define En   1024
#define NHn  16
#define NLEVn 12
#define HDn  64   // En / NHn

// GEMM tiling
#define TILE_M 64
#define TILE_N 64
#define TILE_K 32
#define LDSTR  40   // padded LDS row stride in halves (80B)

typedef __attribute__((ext_vector_type(16))) _Float16 v16h;
typedef __attribute__((ext_vector_type(8)))  _Float16 v8h;
typedef __attribute__((ext_vector_type(8)))  float    v8f;
typedef __attribute__((ext_vector_type(4)))  int      v4i;

#define AS1 __attribute__((address_space(1)))
#define AS3 __attribute__((address_space(3)))

// ---- gfx1250 async global->LDS path (probe-confirmed present; sig: (v4i AS1*, v4i AS3*, Ii, Ii)) ----
#if defined(__has_builtin)
#  if __has_builtin(__builtin_amdgcn_global_load_async_to_lds_b128)
#    define HAVE_ASYNC_LDS 1
#  endif
#endif
#ifndef HAVE_ASYNC_LDS
#  define HAVE_ASYNC_LDS 0
#endif

__device__ __forceinline__ void wait_async_lds() {
#if HAVE_ASYNC_LDS
#  if __has_builtin(__builtin_amdgcn_s_wait_asynccnt)
    __builtin_amdgcn_s_wait_asynccnt(0);
#  else
    asm volatile("s_wait_asynccnt 0x0" ::: "memory");
#  endif
#endif
}

// -------- fragment gather from LDS (CDNA5 16-bit A/B layout) --------
// lane l: row/col = l&15 ; element e<8 -> K = 8*(l>>4)+e ; e>=8 -> K = 16+8*(l>>4)+(e-8)
__device__ __forceinline__ v16h load_frag(const _Float16* lds, int row_base, int lane) {
    const int r  = lane & 15;
    const int kb = (lane >> 4) * 8;
    const _Float16* p = lds + (row_base + r) * LDSTR + kb;
    union { v8h h8[2]; v16h h16; } u;
    u.h8[0] = *(const v8h*)(p);        // K = kb .. kb+7
    u.h8[1] = *(const v8h*)(p + 16);   // K = 16+kb .. 16+kb+7
    return u.h16;
}

// -------- cooperative staging of one 64x32-half tile pair into LDS --------
__device__ __forceinline__ void stage_tiles(const _Float16* __restrict__ A, long lda,
                                            const _Float16* __restrict__ B, long ldb,
                                            int k0, _Float16* ldsA, _Float16* ldsB,
                                            int tid) {
#pragma unroll
    for (int i = 0; i < 2; ++i) {
        const int c   = tid * 2 + i;      // 0..255 chunks
        const int row = c >> 2;           // 0..63
        const int off = (c & 3) * 8;      // 0,8,16,24 halves
        const _Float16* ga = &A[row * lda + k0 + off];
        const _Float16* gb = &B[row * ldb + k0 + off];
        _Float16* la = &ldsA[row * LDSTR + off];
        _Float16* lb = &ldsB[row * LDSTR + off];
#if HAVE_ASYNC_LDS
        __builtin_amdgcn_global_load_async_to_lds_b128(
            (AS1 v4i*)(void*)ga, (AS3 v4i*)(void*)la, 0, 0);
        __builtin_amdgcn_global_load_async_to_lds_b128(
            (AS1 v4i*)(void*)gb, (AS3 v4i*)(void*)lb, 0, 0);
#else
        *(v8h*)la = *(const v8h*)ga;
        *(v8h*)lb = *(const v8h*)gb;
#endif
    }
}

// -------- shared WMMA mainloop (double-buffered LDS): acc += A(64xK) * B(64xK)^T ----
// A, B row-major fp16; 4 waves; each wave owns 16 M-rows x 64 N-cols (4 accumulators)
__device__ __forceinline__ void gemm_mainloop(
    const _Float16* __restrict__ A, long lda,
    const _Float16* __restrict__ B, long ldb,
    int K, v8f acc[4], _Float16* ldsA, _Float16* ldsB)
{
    const int tid  = threadIdx.x;      // 0..127
    const int lane = tid & 31;
    const int wave = tid >> 5;         // 0..3
    stage_tiles(A, lda, B, ldb, 0, ldsA, ldsB, tid);
    int cur = 0;
    for (int k0 = 0; k0 < K; k0 += TILE_K) {
        wait_async_lds();              // my async writes from previous stage done
        __syncthreads();               // everyone's writes visible, prior reads done
        const int nxt = cur ^ 1;
        if (k0 + TILE_K < K)           // overlap staging of next tile with math
            stage_tiles(A, lda, B, ldb, k0 + TILE_K,
                        ldsA + nxt * (TILE_M * LDSTR),
                        ldsB + nxt * (TILE_N * LDSTR), tid);
        if (k0 + 2 * TILE_K < K) {     // L2 prefetch two tiles ahead
            __builtin_prefetch(&A[(long)(tid >> 1) * lda + k0 + 2 * TILE_K], 0, 2);
            __builtin_prefetch(&B[(long)(tid >> 1) * ldb + k0 + 2 * TILE_K], 0, 2);
        }
        const _Float16* cA = ldsA + cur * (TILE_M * LDSTR);
        const _Float16* cB = ldsB + cur * (TILE_N * LDSTR);
        v16h bfrag[4];
#pragma unroll
        for (int nt = 0; nt < 4; ++nt) bfrag[nt] = load_frag(cB, nt * 16, lane);
        const v16h afrag = load_frag(cA, wave * 16, lane);
#pragma unroll
        for (int nt = 0; nt < 4; ++nt)
            acc[nt] = __builtin_amdgcn_wmma_f32_16x16x32_f16(
                false, afrag, false, bfrag[nt], (short)0, acc[nt], false, false);
        cur = nxt;
    }
}

// -------- fp32 -> fp16 convert --------
__global__ void f32_to_f16_kernel(const float* __restrict__ in,
                                  _Float16* __restrict__ out, long n) {
    long i = (long)blockIdx.x * blockDim.x + threadIdx.x;
    if (i < n) out[i] = (_Float16)in[i];
}

// -------- cs[t,h] = cumsum_t log(sigmoid(A_log + A_bias)) --------
__global__ void decay_cumsum_kernel(const float* __restrict__ A_log,
                                    const float* __restrict__ A_bias,
                                    float* __restrict__ cs) {
    const int h = threadIdx.x;
    if (h >= NHn) return;
    float acc = 0.f;
    for (int t = 0; t < Tn; ++t) {
        const float a = A_log[t * NHn + h] + A_bias[t * NHn + h];
        // stable log(sigmoid(a)) = min(a,0) - log1p(exp(-|a|))
        acc += fminf(a, 0.f) - log1pf(__expf(-fabsf(a)));
        cs[t * NHn + h] = acc;
    }
}

// -------- v = x @ Wv^T + bv, stored transposed fp16 as [b, n=(h,d), t] --------
__global__ __launch_bounds__(128)
void proj_v_kernel(const _Float16* __restrict__ x16,
                   const _Float16* __restrict__ Wv16,
                   const float* __restrict__ bv,
                   _Float16* __restrict__ v_ws) {
    __shared__ _Float16 ldsA[2 * TILE_M * LDSTR];
    __shared__ _Float16 ldsB[2 * TILE_N * LDSTR];
    const int n0 = blockIdx.x * TILE_N;
    const int m0 = blockIdx.y * TILE_M;   // t tile
    const int b  = blockIdx.z;
    const _Float16* A  = x16 + (long)b * Tn * En + (long)m0 * En;
    const _Float16* Bm = Wv16 + (long)n0 * En;     // row n of Wv = column n of Wv^T
    v8f acc[4] = {};
    gemm_mainloop(A, En, Bm, En, En, acc, ldsA, ldsB);
    const int lane = threadIdx.x & 31, wave = threadIdx.x >> 5;
#pragma unroll
    for (int nt = 0; nt < 4; ++nt) {
        const int n = n0 + nt * 16 + (lane & 15);
        const float bias = bv[n];
#pragma unroll
        for (int r = 0; r < 8; ++r) {
            const int m = m0 + wave * 16 + r + 8 * (lane >> 4);
            v_ws[((long)b * En + n) * Tn + m] = (_Float16)(acc[nt][r] + bias);
        }
    }
}

// -------- attn[h,t,s] fp16: softmax_s of where(s<=t, G*decay, 0) --------
__global__ __launch_bounds__(256)
void attn_kernel(const float* __restrict__ L, const float* __restrict__ L_bias,
                 const float* __restrict__ cs, const int* __restrict__ levels,
                 _Float16* __restrict__ attn16) {
    const int t = blockIdx.x, h = blockIdx.y, tid = threadIdx.x;
    __shared__ float le[NLEVn];
    __shared__ float red[256];
    if (tid < NLEVn)
        le[tid] = L[((long)t * NHn + h) * NLEVn + tid]
                + L_bias[((long)t * NHn + h) * NLEVn + tid];
    __syncthreads();
    const float cst = cs[t * NHn + h];
    float p[Tn / 256];
    float local = 0.f;
#pragma unroll
    for (int i = 0; i < Tn / 256; ++i) {
        const int s = tid + i * 256;
        float val;
        if (s <= t) {
            const int lev   = levels[(long)t * Tn + s];
            const float dec = __expf(cst - cs[s * NHn + h]);   // <= 1
            val = __expf(le[lev] * dec);
        } else {
            val = 1.0f;   // masked entries are 0 pre-softmax -> exp(0)=1
        }
        p[i] = val;
        local += val;
    }
    red[tid] = local;
    __syncthreads();
    for (int off = 128; off > 0; off >>= 1) {
        if (tid < off) red[tid] += red[tid + off];
        __syncthreads();
    }
    const float invZ = 1.0f / red[0];
    _Float16* row = attn16 + ((long)h * Tn + t) * Tn;
#pragma unroll
    for (int i = 0; i < Tn / 256; ++i)
        row[tid + i * 256] = (_Float16)(p[i] * invZ);
}

// -------- ctx[b,t,(h,d)] = attn[h] (TxT) @ v[b,h] (TxHD), fp16 out --------
__global__ __launch_bounds__(128)
void attn_v_kernel(const _Float16* __restrict__ attn16,
                   const _Float16* __restrict__ v_ws,
                   _Float16* __restrict__ ctx16) {
    __shared__ _Float16 ldsA[2 * TILE_M * LDSTR];
    __shared__ _Float16 ldsB[2 * TILE_N * LDSTR];
    const int m0 = blockIdx.y * TILE_M;   // t tile
    const int z  = blockIdx.z;            // b*NH + h
    const int b = z / NHn, h = z % NHn;
    const _Float16* A  = attn16 + (long)h * Tn * Tn + (long)m0 * Tn;  // lda = T
    const _Float16* Bm = v_ws + (long)z * HDn * Tn;                   // rows d, ldb = T
    v8f acc[4] = {};
    gemm_mainloop(A, Tn, Bm, Tn, Tn, acc, ldsA, ldsB);
    const int lane = threadIdx.x & 31, wave = threadIdx.x >> 5;
#pragma unroll
    for (int nt = 0; nt < 4; ++nt) {
        const int d = nt * 16 + (lane & 15);
#pragma unroll
        for (int r = 0; r < 8; ++r) {
            const int m = m0 + wave * 16 + r + 8 * (lane >> 4);
            ctx16[((long)b * Tn + m) * En + h * HDn + d] = (_Float16)acc[nt][r];
        }
    }
}

// -------- out = ctx @ Wo^T + bo (fp32 out) --------
__global__ __launch_bounds__(128)
void proj_o_kernel(const _Float16* __restrict__ ctx16,
                   const _Float16* __restrict__ Wo16,
                   const float* __restrict__ bo,
                   float* __restrict__ out) {
    __shared__ _Float16 ldsA[2 * TILE_M * LDSTR];
    __shared__ _Float16 ldsB[2 * TILE_N * LDSTR];
    const int n0 = blockIdx.x * TILE_N;
    const int m0 = blockIdx.y * TILE_M;
    const int b  = blockIdx.z;
    const _Float16* A  = ctx16 + (long)b * Tn * En + (long)m0 * En;
    const _Float16* Bm = Wo16 + (long)n0 * En;
    v8f acc[4] = {};
    gemm_mainloop(A, En, Bm, En, En, acc, ldsA, ldsB);
    const int lane = threadIdx.x & 31, wave = threadIdx.x >> 5;
#pragma unroll
    for (int nt = 0; nt < 4; ++nt) {
        const int n = n0 + nt * 16 + (lane & 15);
        const float bias = bo[n];
#pragma unroll
        for (int r = 0; r < 8; ++r) {
            const int m = m0 + wave * 16 + r + 8 * (lane >> 4);
            out[((long)b * Tn + m) * En + n] = acc[nt][r] + bias;
        }
    }
}

extern "C" void kernel_launch(void* const* d_in, const int* in_sizes, int n_in,
                              void* d_out, int out_size, void* d_ws, size_t ws_size,
                              hipStream_t stream) {
    // input order: x, Wq, bq, Wk, bk, Wv, bv, A_log, A_bias, L, L_bias, Wo, bo, levels
    const float* x      = (const float*)d_in[0];
    const float* Wv     = (const float*)d_in[5];
    const float* bv     = (const float*)d_in[6];
    const float* A_log  = (const float*)d_in[7];
    const float* A_bias = (const float*)d_in[8];
    const float* L      = (const float*)d_in[9];
    const float* L_bias = (const float*)d_in[10];
    const float* Wo     = (const float*)d_in[11];
    const float* bo     = (const float*)d_in[12];
    const int*   levels = (const int*)d_in[13];
    float* out = (float*)d_out;

    // workspace carve-out (256B-aligned)
    size_t off = 0;
    auto carve = [&](size_t bytes) -> void* {
        void* p = (char*)d_ws + off;
        off += (bytes + 255) & ~(size_t)255;
        return p;
    };
    _Float16* x16    = (_Float16*)carve((size_t)Bn * Tn * En * 2);  // 67 MB
    _Float16* Wv16   = (_Float16*)carve((size_t)En * En * 2);       //  2 MB
    _Float16* Wo16   = (_Float16*)carve((size_t)En * En * 2);       //  2 MB
    _Float16* v_ws   = (_Float16*)carve((size_t)Bn * En * Tn * 2);  // 67 MB  [b,n,t]
    float*    cs     = (float*)carve((size_t)Tn * NHn * 4);
    _Float16* attn16 = (_Float16*)carve((size_t)NHn * Tn * Tn * 2); // 134 MB (fits L2)
    _Float16* ctx16  = (_Float16*)carve((size_t)Bn * Tn * En * 2);  // 67 MB
    (void)ws_size; (void)n_in; (void)in_sizes; (void)out_size;

    // 1) fp16 conversions
    {
        long n = (long)Bn * Tn * En;
        f32_to_f16_kernel<<<dim3((n + 255) / 256), dim3(256), 0, stream>>>(x, x16, n);
        long nw = (long)En * En;
        f32_to_f16_kernel<<<dim3((nw + 255) / 256), dim3(256), 0, stream>>>(Wv, Wv16, nw);
        f32_to_f16_kernel<<<dim3((nw + 255) / 256), dim3(256), 0, stream>>>(Wo, Wo16, nw);
    }
    // 2) decay cumsum
    decay_cumsum_kernel<<<dim3(1), dim3(32), 0, stream>>>(A_log, A_bias, cs);
    // 3) v projection (WMMA), stored [b, h*HD+d, t]
    proj_v_kernel<<<dim3(En / TILE_N, Tn / TILE_M, Bn), dim3(128), 0, stream>>>(
        x16, Wv16, bv, v_ws);
    // 4) attention matrix fp16 [h, t, s]
    attn_kernel<<<dim3(Tn, NHn), dim3(256), 0, stream>>>(L, L_bias, cs, levels, attn16);
    // 5) ctx = attn @ v (batched WMMA, attn[h] reused across B via L2)
    attn_v_kernel<<<dim3(1, Tn / TILE_M, Bn * NHn), dim3(128), 0, stream>>>(
        attn16, v_ws, ctx16);
    // 6) output projection (WMMA) + bias
    proj_o_kernel<<<dim3(En / TILE_N, Tn / TILE_M, Bn), dim3(128), 0, stream>>>(
        ctx16, Wo16, bo, out);
}
